// Network_85220741087986
// MI455X (gfx1250) — matrix operand order repaired and checked by reading, not compiled, not run on hardware
//
#include <hip/hip_runtime.h>

typedef __bf16 bf16_t;
typedef __attribute__((ext_vector_type(16))) __bf16 v16bf;
typedef __attribute__((ext_vector_type(8)))  float  v8f;
typedef unsigned int u32x4 __attribute__((ext_vector_type(4)));
typedef int          i32x4 __attribute__((ext_vector_type(4)));
typedef int          i32x8 __attribute__((ext_vector_type(8)));

#define MDIM 16384
#define KDIM 1024
#define NOUT 2048
#define GATES 1024

// ---------------- X f32 -> bf16, pre-scaled by layer_scale ----------------
__global__ void convert_x_kernel(const float* __restrict__ X,
                                 bf16_t* __restrict__ Xb,
                                 const float* __restrict__ scale) {
  const float ls = cbrtf(fabsf(scale[0]));           // overall_scale^(1/DEPTH)
  int i = blockIdx.x * blockDim.x + threadIdx.x;     // one float4 per thread
  const float4 x = ((const float4*)X)[i];
  union { bf16_t h[4]; unsigned long long u; } r;
  r.h[0] = (bf16_t)(x.x * ls);
  r.h[1] = (bf16_t)(x.y * ls);
  r.h[2] = (bf16_t)(x.z * ls);
  r.h[3] = (bf16_t)(x.w * ls);
  ((unsigned long long*)Xb)[i] = r.u;
}

// ------------- W = w_top - w_bot, transposed to [NOUT][KDIM] bf16 ----------
__global__ void prep_w_kernel(const float* __restrict__ wpn,
                              bf16_t* __restrict__ Wt) {
  int id = blockIdx.x * blockDim.x + threadIdx.x;    // KDIM*NOUT threads
  int k = id >> 11;          // / NOUT
  int n = id & (NOUT - 1);
  float d = wpn[(size_t)k * NOUT + n] - wpn[(size_t)(k + KDIM) * NOUT + n];
  Wt[(size_t)n * KDIM + k] = (bf16_t)d;
}

// ---- TDM: DMA one [128 rows x 64B] tile into LDS with 16B/row padding ----
// data_size = 1 byte => all dims in bytes. Row in tensor = 2048B (K*2).
// pad_interval = 3 (16 DWORDs = 64B), pad_amount = 3 (4 DWORDs = 16B)
// -> LDS row stride = 80B, matching the bank-conflict-free fragment layout.
__device__ __forceinline__ void tdm_load_tile(const void* gbase, unsigned ldsOff) {
  unsigned long long ga = (unsigned long long)(uintptr_t)gbase;
  u32x4 g0;
  g0[0] = 1u;                                        // count=1, no gather
  g0[1] = ldsOff;                                    // lds_addr [63:32]
  g0[2] = (unsigned)(ga & 0xffffffffu);              // global_addr [95:64]
  g0[3] = (unsigned)((ga >> 32) & 0x1ffffffu)        // global_addr [120:96]
        | (2u << 30);                                // type=2 ("image") [127:126]
  i32x8 g1;
  g1[0] = (1 << 20) | (3 << 22) | (3 << 25);         // pad_en | interval | amount
  g1[1] = (int)((2048u & 0xffffu) << 16);            // tensor_dim0 lo16 @ [63:48]
  g1[2] = (int)(128u << 16);                         // dim0 hi=0 ; tensor_dim1 lo16=128
  g1[3] = (int)(64u  << 16);                         // dim1 hi=0 ; tile_dim0=64B
  g1[4] = 128;                                       // tile_dim1=128 ; tile_dim2=0
  g1[5] = 2048;                                      // tensor_dim0_stride lo32 (bytes)
  g1[6] = 0;
  g1[7] = 0;
  i32x4 z4 = {0, 0, 0, 0};
#if defined(__clang_major__) && (__clang_major__ >= 23)
  i32x8 z8 = {0, 0, 0, 0, 0, 0, 0, 0};
  __builtin_amdgcn_tensor_load_to_lds(g0, g1, z4, z4, z8, 0);
#else
  __builtin_amdgcn_tensor_load_to_lds(g0, g1, z4, z4, 0);
#endif
}

// --------------------------- fused GEMM + gate act -------------------------
template <bool LAST>
__global__ __launch_bounds__(256)
void gemm_act_kernel(const bf16_t* __restrict__ A,    // [MDIM][KDIM] bf16 (pre-scaled)
                     const bf16_t* __restrict__ Bw,   // [NOUT][KDIM] bf16
                     const float*  __restrict__ bias, // [NOUT]
                     const float*  __restrict__ aprm, // [GATES*9]
                     const float*  __restrict__ scale,
                     bf16_t*       __restrict__ nxt,  // [MDIM][GATES] bf16 (if !LAST)
                     const float*  __restrict__ obias,// [GATES]        (if LAST)
                     float*        __restrict__ outF) // [MDIM][GATES]  (if LAST)
{
  constexpr int BM = 128, BN = 128, BK = 32;
  constexpr int ROWB  = BK * 2 + 16;                 // 80B row stride (bank-safe)
  constexpr int TILEB = BM * ROWB;                   // 10240B per tile
  constexpr int BUFB  = 2 * TILEB;                   // A+B per buffer
  // Single shared object -> group-segment offset 0; TDM lds_addr = byte offset.
  __shared__ __align__(16) unsigned char lds_all[2 * BUFB];  // 40 KB, double-buffered

  const int tid  = threadIdx.x;
  const int lane = tid & 31;
  const int wave = tid >> 5;
  const int wm   = wave & 3;     // 4 waves along M -> 32 rows each
  const int wn   = wave >> 2;    // 2 waves along N -> 64 cols each
  const int m0   = blockIdx.y * BM;
  const int n0   = blockIdx.x * BN;
  const int l15  = lane & 15;
  const bool hi  = lane >= 16;

  v8f acc[2][4] = {};

  const bf16_t* Abase = A  + (size_t)m0 * KDIM;
  const bf16_t* Bbase = Bw + (size_t)n0 * KDIM;

  // prologue: DMA first tiles into buffer 0
  if (wave == 0) {
    tdm_load_tile(Abase, 0);
    tdm_load_tile(Bbase, TILEB);
  }

  int cur = 0;
  for (int k0 = 0; k0 < KDIM; k0 += BK) {
    const bool more = (k0 + BK) < KDIM;
    if (wave == 0) {
      if (more) {
        // overlap: DMA next tiles while we compute on current buffer
        tdm_load_tile(Abase + (k0 + BK), (unsigned)((cur ^ 1) * BUFB));
        tdm_load_tile(Bbase + (k0 + BK), (unsigned)((cur ^ 1) * BUFB + TILEB));
        __builtin_amdgcn_s_wait_tensorcnt(2);   // current tile's 2 loads done
      } else {
        __builtin_amdgcn_s_wait_tensorcnt(0);
      }
    }
    __syncthreads();                            // publish: current buffer ready

    union Frag { uint4 q[2]; v16bf v; };
    Frag fa[2], fb[4];
    // A 16x32 bf16 frag: lane<16 -> K 0..7 & 16..23 ; lane>=16 -> K 8..15 & 24..31
    const int abase = cur * BUFB + (hi ? 16 : 0);
#pragma unroll
    for (int mt = 0; mt < 2; ++mt) {
      int off = abase + (wm * 32 + mt * 16 + l15) * ROWB;
      fa[mt].q[0] = *(const uint4*)(&lds_all[off]);
      fa[mt].q[1] = *(const uint4*)(&lds_all[off + 32]);
    }
    // B 32x16 bf16 frag: lane<16 -> K 0..7 & 16..23 ; lane>=16 -> K 8..15 & 24..31
    const int bbase = cur * BUFB + TILEB + (hi ? 16 : 0);
#pragma unroll
    for (int nt = 0; nt < 4; ++nt) {
      int off = bbase + (wn * 64 + nt * 16 + l15) * ROWB;
      fb[nt].q[0] = *(const uint4*)(&lds_all[off]);
      fb[nt].q[1] = *(const uint4*)(&lds_all[off + 32]);
    }
#pragma unroll
    for (int mt = 0; mt < 2; ++mt)
#pragma unroll
      for (int nt = 0; nt < 4; ++nt)
        acc[mt][nt] = __builtin_amdgcn_wmma_f32_16x16x32_bf16(
            false, fa[mt].v, false, fb[nt].v,
            (short)0, acc[mt][nt], false, false);

    __syncthreads();              // all waves done reading before DMA overwrites
    cur ^= 1;
  }

  // ------------- epilogue: bias + 2-axis multilinear gate activation -------
  const float ls = cbrtf(fabsf(scale[0]));
#pragma unroll
  for (int nt = 0; nt < 4; ++nt) {
    int n = n0 + wn * 64 + nt * 16 + l15;            // Y column (0..2047)
    float bn = bias[n];
    int g = n >> 1;                                   // gate index
    const float* p = &aprm[(size_t)g * 9];
    float p0 = p[0], p1 = p[1], p2 = p[2], p3 = p[3], p4 = p[4],
          p5 = p[5], p6 = p[6], p7 = p[7], p8 = p[8];
    const bool even = ((l15 & 1) == 0);
    float ob = 0.f;
    if (LAST) ob = obias[g];
#pragma unroll
    for (int mt = 0; mt < 2; ++mt) {
#pragma unroll
      for (int r = 0; r < 8; ++r) {
        float y = acc[mt][nt][r] + bn;
        float o = __shfl_xor(y, 1, 32);              // partner column of the pair
        float x0 = even ? y : o;
        float x1 = even ? o : y;
        x0 = fminf(1.f, fmaxf(-1.f, x0));
        x1 = fminf(1.f, fmaxf(-1.f, x1));
        float l0 = (x0 < 0.f) ? -1.f : 0.f;
        float l1 = (x1 < 0.f) ? -1.f : 0.f;
        float f0 = x0 - l0;
        float f1 = x1 - l1;
        bool s0 = (l0 == 0.f);
        bool s1 = (l1 == 0.f);
        // vertex params: base = s0 + 3*s1 ; a00=p[base],a10=p[base+1],a01=p[base+3],a11=p[base+4]
        float a00 = s1 ? (s0 ? p4 : p3) : (s0 ? p1 : p0);
        float a10 = s1 ? (s0 ? p5 : p4) : (s0 ? p2 : p1);
        float a01 = s1 ? (s0 ? p7 : p6) : (s0 ? p4 : p3);
        float a11 = s1 ? (s0 ? p8 : p7) : (s0 ? p5 : p4);
        float outv = (1.f - f0) * ((1.f - f1) * a00 + f1 * a01) +
                     f0 * ((1.f - f1) * a10 + f1 * a11);
        if (even) {
          int row = m0 + wm * 32 + mt * 16 + r + (hi ? 8 : 0);
          if (LAST) outF[(size_t)row * GATES + g] = outv + ob;
          else      nxt[(size_t)row * GATES + g]  = (bf16_t)(outv * ls);
        }
      }
    }
  }
}

// ------------------------------- launcher ---------------------------------
extern "C" void kernel_launch(void* const* d_in, const int* in_sizes, int n_in,
                              void* d_out, int out_size, void* d_ws, size_t ws_size,
                              hipStream_t stream) {
  (void)in_sizes; (void)n_in; (void)out_size; (void)ws_size;
  const float* X  = (const float*)d_in[0];
  const float* w0 = (const float*)d_in[1];
  const float* b0 = (const float*)d_in[2];
  const float* a0 = (const float*)d_in[3];
  const float* w1 = (const float*)d_in[4];
  const float* b1 = (const float*)d_in[5];
  const float* a1 = (const float*)d_in[6];
  const float* w2 = (const float*)d_in[7];
  const float* b2 = (const float*)d_in[8];
  const float* a2 = (const float*)d_in[9];
  const float* sc = (const float*)d_in[10];
  const float* ob = (const float*)d_in[11];
  float* out = (float*)d_out;

  char* ws = (char*)d_ws;
  const size_t XB = (size_t)MDIM * KDIM * sizeof(bf16_t);  // 32 MB
  const size_t WB = (size_t)NOUT * KDIM * sizeof(bf16_t);  //  4 MB
  bf16_t* Xb0 = (bf16_t*)(ws);
  bf16_t* Xb1 = (bf16_t*)(ws + XB);
  bf16_t* W0  = (bf16_t*)(ws + 2 * XB);
  bf16_t* W1  = (bf16_t*)(ws + 2 * XB + WB);
  bf16_t* W2  = (bf16_t*)(ws + 2 * XB + 2 * WB);

  convert_x_kernel<<<dim3((MDIM * KDIM / 4) / 256), 256, 0, stream>>>(X, Xb0, sc);
  prep_w_kernel<<<dim3((KDIM * NOUT) / 256), 256, 0, stream>>>(w0, W0);
  prep_w_kernel<<<dim3((KDIM * NOUT) / 256), 256, 0, stream>>>(w1, W1);
  prep_w_kernel<<<dim3((KDIM * NOUT) / 256), 256, 0, stream>>>(w2, W2);

  dim3 grid(NOUT / 128, MDIM / 128);   // (16, 128)
  gemm_act_kernel<false><<<grid, 256, 0, stream>>>(Xb0, W0, b0, a0, sc, Xb1, nullptr, nullptr);
  gemm_act_kernel<false><<<grid, 256, 0, stream>>>(Xb1, W1, b1, a1, sc, Xb0, nullptr, nullptr);
  gemm_act_kernel<true ><<<grid, 256, 0, stream>>>(Xb0, W2, b2, a2, sc, nullptr, ob, out);
}
